// UpdateUnit_39067022524700
// MI455X (gfx1250) — matrix operand order repaired and checked by reading, not compiled, not run on hardware
//
#include <hip/hip_runtime.h>
#include <hip/hip_bf16.h>

typedef __attribute__((ext_vector_type(16))) _Float16 v16h;
typedef __attribute__((ext_vector_type(8)))  float    v8f;

#define D_DIM 64

// ---------------------------------------------------------------------------
// Wave-level helper: accumulate a 16x64 tile of Y = X[16 rows,64] @ W^T
// into c[4] (four 16x16 f32 fragments), using v_wmma_f32_16x16x32_f16.
//   A (16x32 f16) layout: lane L holds row m=L&15; VGPR v -> K = (v>>2)*16 + (L>>4)*8 + (v&3)*2
//   B (32x16 f16) layout: lane L holds col n=L&15; VGPR v -> K = (v>>2)*16 + (L>>4)*8 + (v&3)*2
//   C/D (16x16 f32):      lane L holds col n=L&15, rows m = 8*(L>>4)+r in VGPR r
// ---------------------------------------------------------------------------
__device__ __forceinline__ void tile_gemm_acc(const float* __restrict__ Xrow,
                                              const float* __restrict__ W,
                                              int lane, v8f c[4]) {
  const int half = lane >> 4;
  const int nloc = lane & 15;
#pragma unroll
  for (int s = 0; s < 2; ++s) {
    v16h a;
#pragma unroll
    for (int v = 0; v < 8; ++v) {
      const int k = s * 32 + (v >> 2) * 16 + half * 8 + (v & 3) * 2;
      const float* xp = Xrow + (size_t)nloc * D_DIM + k;   // A row = lane&15
      a[2 * v]     = (_Float16)xp[0];
      a[2 * v + 1] = (_Float16)xp[1];
    }
#pragma unroll
    for (int t = 0; t < 4; ++t) {
      v16h b;
#pragma unroll
      for (int v = 0; v < 8; ++v) {
        const int k = s * 32 + (v >> 2) * 16 + half * 8 + (v & 3) * 2;
        const float* wp = W + (size_t)(t * 16 + nloc) * D_DIM + k;  // Y[:,n] uses W row n
        b[2 * v]     = (_Float16)wp[0];
        b[2 * v + 1] = (_Float16)wp[1];
      }
      c[t] = __builtin_amdgcn_wmma_f32_16x16x32_f16(
          /*neg_a=*/false, a, /*neg_b=*/false, b,
          /*c_mod=*/(short)0, c[t], /*reuse_a=*/false, /*reuse_b=*/false);
    }
  }
}

// ---------------------------------------------------------------------------
// Kernel 1: Y = X @ W^T  (message transform), one wave per 16-row tile.
// ---------------------------------------------------------------------------
__global__ void uu_gemm_xwT(const float* __restrict__ X, const float* __restrict__ W,
                            float* __restrict__ Y, int M) {
  const int wave    = blockIdx.x * (blockDim.x >> 5) + (threadIdx.x >> 5);
  const int rowbase = wave * 16;
  if (rowbase >= M) return;                      // wave-uniform: EXEC stays all-1s
  const int lane = threadIdx.x & 31;
  const int half = lane >> 4;
  const int nloc = lane & 15;

  v8f c[4] = {};
  tile_gemm_acc(X + (size_t)rowbase * D_DIM, W, lane, c);

#pragma unroll
  for (int t = 0; t < 4; ++t)
#pragma unroll
    for (int r = 0; r < 8; ++r) {
      const int m = half * 8 + r;
      Y[(size_t)(rowbase + m) * D_DIM + t * 16 + nloc] = c[t][r];
    }
}

// ---------------------------------------------------------------------------
// Kernel 2: edge scatter  acc[rows[e]] += vals[e] * Xsrc[cols[e]]; deg[rows[e]] += vals[e]
// 32 lanes per edge, 2 floats per lane.
// ---------------------------------------------------------------------------
__global__ void uu_scatter_edges(const int* __restrict__ rows, const int* __restrict__ cols,
                                 const float* __restrict__ vals,
                                 const float* __restrict__ Xsrc,
                                 float* __restrict__ acc, float* __restrict__ deg, int E) {
  const long long tid = (long long)blockIdx.x * blockDim.x + threadIdx.x;
  const long long e   = tid >> 5;
  const int lane      = (int)(tid & 31);
  if (e >= E) return;
  const int   dst = rows[e];
  const int   src = cols[e];
  const float w   = vals[e];
  const float* xs = Xsrc + (size_t)src * D_DIM;
  float*       ap = acc  + (size_t)dst * D_DIM;
  const int d = lane * 2;
  atomicAdd(&ap[d],     w * xs[d]);
  atomicAdd(&ap[d + 1], w * xs[d + 1]);
  if (lane == 0) atomicAdd(&deg[dst], w);
}

// ---------------------------------------------------------------------------
// Kernel 3: finalize.  delta = relu(X@W^T + bias + acc) * (deg>0);
//           out = X + delta; accumulate sum(delta^2) and mask count.
// ---------------------------------------------------------------------------
__global__ void uu_finalize(const float* __restrict__ X, const float* __restrict__ W,
                            const float* __restrict__ bias,
                            const float* __restrict__ acc, const float* __restrict__ deg,
                            float* __restrict__ out, float* __restrict__ sums, int M) {
  const int wave    = blockIdx.x * (blockDim.x >> 5) + (threadIdx.x >> 5);
  const int rowbase = wave * 16;
  if (rowbase >= M) return;
  const int lane = threadIdx.x & 31;
  const int half = lane >> 4;
  const int nloc = lane & 15;

  v8f c[4] = {};
  tile_gemm_acc(X + (size_t)rowbase * D_DIM, W, lane, c);

  float d2 = 0.0f;
#pragma unroll
  for (int t = 0; t < 4; ++t) {
    const float bn = bias[t * 16 + nloc];
#pragma unroll
    for (int r = 0; r < 8; ++r) {
      const int    m   = half * 8 + r;
      const size_t idx = (size_t)(rowbase + m) * D_DIM + t * 16 + nloc;
      float v = c[t][r] + bn + acc[idx];
      v = fmaxf(v, 0.0f);
      const float msk = (deg[rowbase + m] > 0.0f) ? 1.0f : 0.0f;
      v *= msk;
      d2 += v * v;
      out[idx] = X[idx] + v;
    }
  }

  // mask count: only the n==0 lanes (L=0 and L=16) count their 8 rows -> each row once
  float cnt = 0.0f;
  if (nloc == 0) {
#pragma unroll
    for (int r = 0; r < 8; ++r)
      cnt += (deg[rowbase + half * 8 + r] > 0.0f) ? 1.0f : 0.0f;
  }

  // wave32 reduction
#pragma unroll
  for (int off = 16; off > 0; off >>= 1) {
    d2  += __shfl_down(d2,  off, 32);
    cnt += __shfl_down(cnt, off, 32);
  }
  if (lane == 0) {
    atomicAdd(&sums[0], d2);
    atomicAdd(&sums[1], cnt);
  }
}

// ---------------------------------------------------------------------------
// Kernel 4: delta_norm = s_u/c_u + s_i/c_i
// ---------------------------------------------------------------------------
__global__ void uu_write_norm(const float* __restrict__ scal, float* __restrict__ dst) {
  if (threadIdx.x == 0 && blockIdx.x == 0)
    dst[0] = scal[0] / scal[1] + scal[2] / scal[3];
}

// ---------------------------------------------------------------------------
extern "C" void kernel_launch(void* const* d_in, const int* in_sizes, int n_in,
                              void* d_out, int out_size, void* d_ws, size_t ws_size,
                              hipStream_t stream) {
  const float* user_embs = (const float*)d_in[0];
  const float* item_embs = (const float*)d_in[1];
  const int*   i2u_rows  = (const int*)d_in[2];
  const int*   i2u_cols  = (const int*)d_in[3];
  const float* i2u_vals  = (const float*)d_in[4];
  const int*   u2i_rows  = (const int*)d_in[5];
  const int*   u2i_cols  = (const int*)d_in[6];
  const float* u2i_vals  = (const float*)d_in[7];
  const float* W_uu      = (const float*)d_in[8];
  const float* b_uu      = (const float*)d_in[9];
  const float* W_ii      = (const float*)d_in[10];
  const float* b_ii      = (const float*)d_in[11];
  const float* W_ui      = (const float*)d_in[12];
  const float* W_iu      = (const float*)d_in[13];

  const int NU = in_sizes[0] / D_DIM;
  const int NI = in_sizes[1] / D_DIM;
  const int E  = in_sizes[2];

  float* out = (float*)d_out;
  // d_out doubles as scratch for the transformed embeddings: written by the
  // transform GEMMs, fully consumed by the scatter pass, then overwritten by
  // the finalize pass. Deterministic per call.
  float* Xu = out;                       // [NU,64]  user_embs @ W_ui^T
  float* Xi = out + (size_t)NU * D_DIM;  // [NI,64]  item_embs @ W_iu^T

  // Workspace layout (zeroed each call)
  float* ws    = (float*)d_ws;
  float* scal  = ws;                               // [0]=s_u [1]=c_u [2]=s_i [3]=c_i
  float* deg_u = ws + 16;                          // NU
  float* deg_i = deg_u + NU;                       // NI
  float* acc_u = deg_i + NI;                       // NU*64
  float* acc_i = acc_u + (size_t)NU * D_DIM;       // NI*64
  const size_t ws_used =
      (16 + (size_t)NU + NI + (size_t)(NU + NI) * D_DIM) * sizeof(float);
  hipMemsetAsync(d_ws, 0, ws_used, stream);

  const dim3 blk(256);
  const int wavesPerBlk = 256 / 32;
  const int ublocks = ((NU + 15) / 16 + wavesPerBlk - 1) / wavesPerBlk;
  const int iblocks = ((NI + 15) / 16 + wavesPerBlk - 1) / wavesPerBlk;

  // 1) message transforms (WMMA)
  uu_gemm_xwT<<<ublocks, blk, 0, stream>>>(user_embs, W_ui, Xu, NU);
  uu_gemm_xwT<<<iblocks, blk, 0, stream>>>(item_embs, W_iu, Xi, NI);

  // 2) sparse scatter-add (consumes Xu/Xi in d_out)
  const long long sthreads = (long long)E * 32;
  const int sblocks = (int)((sthreads + 255) / 256);
  uu_scatter_edges<<<sblocks, blk, 0, stream>>>(i2u_rows, i2u_cols, i2u_vals,
                                                Xi, acc_u, deg_u, E);
  uu_scatter_edges<<<sblocks, blk, 0, stream>>>(u2i_rows, u2i_cols, u2i_vals,
                                                Xu, acc_i, deg_i, E);

  // 3) finalize (WMMA self-GEMM + bias + messages + relu + mask + residual)
  uu_finalize<<<ublocks, blk, 0, stream>>>(user_embs, W_uu, b_uu, acc_u, deg_u,
                                           out, scal, NU);
  uu_finalize<<<iblocks, blk, 0, stream>>>(item_embs, W_ii, b_ii, acc_i, deg_i,
                                           out + (size_t)NU * D_DIM, scal + 2, NI);

  // 4) scalar
  uu_write_norm<<<1, 1, 0, stream>>>(scal, out + (size_t)(NU + NI) * D_DIM);
}
